// Compositional_Attention_3367254360493
// MI455X (gfx1250) — matrix operand order repaired and checked by reading, not compiled, not run on hardware
//
#include <hip/hip_runtime.h>
#include <hip/hip_bf16.h>
#include <cstdint>

// ---- problem constants --------------------------------------------------
#define B_   4
#define S_   2048
#define DIM_ 768
#define H_   12
#define HD_  64
#define QK_  32
#define R_   8
#define EPS_ 1e-5f

typedef __bf16 bf16_t;
typedef __attribute__((ext_vector_type(16))) __bf16 v16bf;
typedef __attribute__((ext_vector_type(8)))  __bf16 v8bf;
typedef __attribute__((ext_vector_type(8)))  float  v8f;

static __device__ __forceinline__ v8f wmma_bf16(v16bf a, v16bf b, v8f c) {
  // (neg_a, A, neg_b, B, c_mod, C, reuse_a, reuse_b)
  return __builtin_amdgcn_wmma_f32_16x16x32_bf16(false, a, false, b, (short)0, c,
                                                 false, false);
}

// A fragment 16(M)x32(K), src row-major (ld elements).
// ISA layout: lanes 0-15 (m=lane): K = 0..7 & 16..23 ; lanes 16-31: K = 8..15 & 24..31
static __device__ __forceinline__ v16bf load_frag_a(const bf16_t* src, int ld) {
  int lane = threadIdx.x & 31;
  const bf16_t* p = src + (size_t)(lane & 15) * ld + (lane >> 4) * 8;
  v16bf f;
  *(v8bf*)&f         = *(const v8bf*)(p);
  *(((v8bf*)&f) + 1) = *(const v8bf*)(p + 16);
  return f;
}

// B fragment 32(K)x16(N), src = B^T row-major [N][K] (ld elements).
// ISA layout: lanes 0-15 (n=lane): K = 0..7 & 16..23 ; lanes 16-31: K = 8..15 & 24..31
static __device__ __forceinline__ v16bf load_frag_b(const bf16_t* srcT, int ld) {
  int lane = threadIdx.x & 31;
  const bf16_t* p = srcT + (size_t)(lane & 15) * ld + (lane >> 4) * 8;
  v16bf f;
  *(v8bf*)&f         = *(const v8bf*)(p);
  *(((v8bf*)&f) + 1) = *(const v8bf*)(p + 16);
  return f;
}

// ---- cross-lane reductions within 16-lane groups via v_permlane16_b32 ---
static __device__ __forceinline__ float px16(float v, unsigned s0, unsigned s1) {
  int i = __float_as_int(v);
  int r = __builtin_amdgcn_permlane16(i, i, (int)s0, (int)s1, false, false);
  return __int_as_float(r);
}
static __device__ __forceinline__ float red16_max(float v) {
  v = fmaxf(v, px16(v, 0x67452301u, 0xEFCDAB89u));  // xor 1
  v = fmaxf(v, px16(v, 0x54761032u, 0xDCFE98BAu));  // xor 2
  v = fmaxf(v, px16(v, 0x32107654u, 0xBA98FEDCu));  // xor 4
  v = fmaxf(v, px16(v, 0xFEDCBA98u, 0x76543210u));  // xor 8
  return v;
}
static __device__ __forceinline__ float red16_add(float v) {
  v += px16(v, 0x67452301u, 0xEFCDAB89u);
  v += px16(v, 0x54761032u, 0xDCFE98BAu);
  v += px16(v, 0x32107654u, 0xBA98FEDCu);
  v += px16(v, 0xFEDCBA98u, 0x76543210u);
  return v;
}
static __device__ __forceinline__ float red32_add(float v) {
  v += __shfl_xor(v, 16, 32);  // cross-half once, then VALU permlanes
  return red16_add(v);
}

// ---- async global->LDS (CDNA5 ASYNCcnt path) ----------------------------
static __device__ __forceinline__ void async_cp32(unsigned lds, uint64_t ga) {
  asm volatile("global_load_async_to_lds_b128 %0, %1, off" ::"v"(lds), "v"(ga)
               : "memory");
  asm volatile("global_load_async_to_lds_b128 %0, %1, off" ::"v"(lds + 16u),
               "v"(ga + 16u)
               : "memory");
}
static __device__ __forceinline__ void wait_async0() {
  asm volatile("s_wait_asynccnt 0" ::: "memory");
}

// ---- LayerNorm: one wave32 per 768-wide row, output bf16 ----------------
__global__ __launch_bounds__(256) void ln_kernel(const float* __restrict__ x,
                                                 const float* __restrict__ g,
                                                 const float* __restrict__ b,
                                                 bf16_t* __restrict__ out) {
  int wid = threadIdx.x >> 5, lane = threadIdx.x & 31;
  int row = blockIdx.x * 8 + wid;
  const float* xr = x + (size_t)row * DIM_;
  float v[24];
  float s = 0.f;
#pragma unroll
  for (int i = 0; i < 24; ++i) { v[i] = xr[lane + i * 32]; s += v[i]; }
  s = red32_add(s);
  float mean = s * (1.f / DIM_);
  float vs = 0.f;
#pragma unroll
  for (int i = 0; i < 24; ++i) { float d = v[i] - mean; vs += d * d; }
  vs = red32_add(vs);
  float rstd = rsqrtf(vs * (1.f / DIM_) + EPS_);
  bf16_t* orow = out + (size_t)row * DIM_;
#pragma unroll
  for (int i = 0; i < 24; ++i) {
    int c = lane + i * 32;
    orow[c] = (bf16_t)((v[i] - mean) * rstd * g[c] + b[c]);
  }
}

// ---- weight pack: Wt[n*K+k] = W[k*N+n] * scale, fp32 -> bf16 ------------
__global__ void packT_kernel(const float* __restrict__ W, bf16_t* __restrict__ Wt,
                             int K, int N, float scale) {
  size_t gid = (size_t)blockIdx.x * 256 + threadIdx.x;
  if (gid >= (size_t)K * N) return;
  int k = (int)(gid % K), n = (int)(gid / K);
  Wt[gid] = (bf16_t)(W[(size_t)k * N + n] * scale);
}

__global__ void packb_kernel(const float* __restrict__ src, float* __restrict__ dst,
                             int n, float scale) {
  int i = blockIdx.x * 256 + threadIdx.x;
  if (i < n) dst[i] = src[i] * scale;
}

// ---- generic bf16 WMMA GEMM: C = A[MxK] * B[KxN] (B given transposed) ---
// Double-buffered LDS pipeline fed by async global->LDS copies.
#define BM 128
#define BN 128
#define BKT 32
#define ASTR 40  // padded LDS row stride (bf16 elems); 80B keeps b128 alignment

enum { EPI_BF16 = 0, EPI_BF16_RELU = 1, EPI_F32_RES = 2 };

template <int EPI>
__global__ __launch_bounds__(256) void gemm_kernel(
    const bf16_t* __restrict__ A, const bf16_t* __restrict__ Bt, int M, int N,
    int K, const float* __restrict__ bias, const float* __restrict__ resid,
    void* __restrict__ out) {
  __shared__ __align__(16) bf16_t sA[2][BM * ASTR];
  __shared__ __align__(16) bf16_t sB[2][BN * ASTR];
  const unsigned BUFB = (unsigned)(BM * ASTR * sizeof(bf16_t));  // 10240B

  int t = threadIdx.x;
  int bm = blockIdx.y * BM, bn = blockIdx.x * BN;
  int wid = t >> 5, lane = t & 31;
  int wm = wid >> 1, wn = wid & 1;   // 4x2 wave grid -> 32x64 tile per wave
  int lrow = t >> 1, lhalf = t & 1;  // cooperative tile loads (32B per thread)
  const bf16_t* gA = A + (size_t)(bm + lrow) * K + lhalf * 16;
  const bf16_t* gB = Bt + (size_t)(bn + lrow) * K + lhalf * 16;
  unsigned la0 = (unsigned)(uintptr_t)(&sA[0][0] + lrow * ASTR + lhalf * 16);
  unsigned lb0 = (unsigned)(uintptr_t)(&sB[0][0] + lrow * ASTR + lhalf * 16);

  v8f zf = {0.f, 0.f, 0.f, 0.f, 0.f, 0.f, 0.f, 0.f};
  v8f acc[2][4];
#pragma unroll
  for (int i = 0; i < 2; ++i)
#pragma unroll
    for (int j = 0; j < 4; ++j) acc[i][j] = zf;

  // prologue: async-stage tile 0 into buffer 0
  async_cp32(la0, (uint64_t)(uintptr_t)gA);
  async_cp32(lb0, (uint64_t)(uintptr_t)gB);
  wait_async0();
  __syncthreads();

  int buf = 0;
  for (int kk = 0; kk < K; kk += BKT) {
    int nbuf = buf ^ 1;
    if (kk + BKT < K) {  // overlap: async-stage next tile while computing
      async_cp32(la0 + (unsigned)nbuf * BUFB,
                 (uint64_t)(uintptr_t)(gA + kk + BKT));
      async_cp32(lb0 + (unsigned)nbuf * BUFB,
                 (uint64_t)(uintptr_t)(gB + kk + BKT));
    }
    v16bf af0 = load_frag_a(sA[buf] + (size_t)(wm * 32) * ASTR, ASTR);
    v16bf af1 = load_frag_a(sA[buf] + (size_t)(wm * 32 + 16) * ASTR, ASTR);
#pragma unroll
    for (int nt = 0; nt < 4; ++nt) {
      v16bf bfr = load_frag_b(sB[buf] + (size_t)(wn * 64 + nt * 16) * ASTR, ASTR);
      acc[0][nt] = wmma_bf16(af0, bfr, acc[0][nt]);
      acc[1][nt] = wmma_bf16(af1, bfr, acc[1][nt]);
    }
    wait_async0();
    __syncthreads();
    buf = nbuf;
  }

  int nhalf = lane >> 4, nin = lane & 15;
#pragma unroll
  for (int mt = 0; mt < 2; ++mt)
#pragma unroll
    for (int nt = 0; nt < 4; ++nt) {
      int col = bn + wn * 64 + nt * 16 + nin;
      float bv = bias[col];
#pragma unroll
      for (int j = 0; j < 8; ++j) {
        int row = bm + wm * 32 + mt * 16 + j + nhalf * 8;
        float vv = acc[mt][nt][j] + bv;
        if (EPI == EPI_BF16_RELU) vv = fmaxf(vv, 0.f);
        if (EPI == EPI_F32_RES) {
          ((float*)out)[(size_t)row * N + col] = vv + resid[(size_t)row * N + col];
        } else {
          ((bf16_t*)out)[(size_t)row * N + col] = (bf16_t)vv;
        }
      }
    }
}

// ---- V re-layout: vT[b][c][s] = qkv[b*S+s][1536+c]  (feature-major) -----
__global__ void vt_kernel(const bf16_t* __restrict__ qkv, bf16_t* __restrict__ vT) {
  size_t gid = (size_t)blockIdx.x * 256 + threadIdx.x;
  int s = (int)(gid & 2047);
  int c = (int)((gid >> 11) & 511);
  int b = (int)(gid >> 20);
  vT[gid] = qkv[((size_t)(b * S_ + s)) * 2432 + 1536 + c];
}

// ---- qv_dot[b*S+s][h] = bs + sum_j qv[...,j]*ws_q[j] --------------------
__global__ void qvdot_kernel(const bf16_t* __restrict__ qkv,
                             const float* __restrict__ Ws,
                             const float* __restrict__ bs,
                             float* __restrict__ qvd) {
  int gid = blockIdx.x * 256 + threadIdx.x;
  if (gid >= B_ * S_ * H_) return;
  int h = gid % H_;
  int sb = gid / H_;  // b*S + s
  const bf16_t* p = qkv + (size_t)sb * 2432 + 2048 + h * 32;
  float a = bs[0];
#pragma unroll
  for (int j = 0; j < 32; ++j) a += (float)p[j] * Ws[j];
  qvd[gid] = a;
}

// ---- fused flash attention + compositional mixing -----------------------
// grid (S/32, H, B), 256 threads = 8 waves. 4 waves share 16 queries; each
// wave owns 2 of the 8 rules (64 acc VGPRs), redundant QK^T, no barriers in
// the key loop. Row-sum of P done with a ones-matrix WMMA; row-max via
// v_permlane16_b32. Epilogue fuses comp / rule-softmax / weighted mix.
__global__ __launch_bounds__(256) void attn_kernel(
    const bf16_t* __restrict__ qkv, const bf16_t* __restrict__ vT,
    const float* __restrict__ qvd, const float* __restrict__ Ws,
    bf16_t* __restrict__ obuf, float* __restrict__ prev) {
  __shared__ __align__(16) bf16_t sP[8][16 * 32];  // per-wave P tile (C->A relayout)
  __shared__ float sComp[2][16][8];
  __shared__ float sO[2][4][16][64];

  int t = threadIdx.x, lane = t & 31, wid = t >> 5;
  int qg = wid >> 2, wr = wid & 3, r0 = wr * 2;
  int nhalf = lane >> 4, nin = lane & 15;
  int b = blockIdx.z, h = blockIdx.y;
  int q0 = blockIdx.x * 32 + qg * 16;

  const bf16_t* qbase = qkv + ((size_t)(b * S_ + q0)) * 2432 + h * 64;
  const bf16_t* kbase = qkv + ((size_t)b * S_) * 2432 + 768 + h * 64;
  const bf16_t* vb = vT + (size_t)b * 512 * 2048;

  v16bf aq0 = load_frag_a(qbase, 2432);       // hd 0..31 (1/sqrt(HD) folded in Wq)
  v16bf aq1 = load_frag_a(qbase + 32, 2432);  // hd 32..63

  v16bf ones;
#pragma unroll
  for (int i = 0; i < 16; ++i) ones[i] = (bf16_t)1.0f;

  v8f zf = {0.f, 0.f, 0.f, 0.f, 0.f, 0.f, 0.f, 0.f};
  v8f acc[2][4];
#pragma unroll
  for (int i = 0; i < 2; ++i)
#pragma unroll
    for (int j = 0; j < 4; ++j) acc[i][j] = zf;
  float rm[8], rs[8];
#pragma unroll
  for (int j = 0; j < 8; ++j) { rm[j] = -3e38f; rs[j] = 0.f; }

  bf16_t* ps = sP[wid];

  for (int kk = 0; kk < S_; kk += 32) {
    const bf16_t* kb = kbase + (size_t)kk * 2432;
    v8f s0 = zf, s1 = zf;
    s0 = wmma_bf16(aq0, load_frag_b(kb, 2432), s0);
    s0 = wmma_bf16(aq1, load_frag_b(kb + 32, 2432), s0);
    s1 = wmma_bf16(aq0, load_frag_b(kb + (size_t)16 * 2432, 2432), s1);
    s1 = wmma_bf16(aq1, load_frag_b(kb + (size_t)16 * 2432 + 32, 2432), s1);

#pragma unroll
    for (int j = 0; j < 8; ++j) {
      float tm = red16_max(fmaxf(s0[j], s1[j]));  // pure-VALU row max
      float nm = fmaxf(rm[j], tm);
      float sc = __expf(rm[j] - nm);
      float p0 = __expf(s0[j] - nm);
      float p1 = __expf(s1[j] - nm);
      rs[j] *= sc;
      rm[j] = nm;
      int m = j + nhalf * 8;
      ps[m * 32 + nin] = (bf16_t)p0;
      ps[m * 32 + 16 + nin] = (bf16_t)p1;
#pragma unroll
      for (int rw = 0; rw < 2; ++rw)
#pragma unroll
        for (int ct = 0; ct < 4; ++ct) acc[rw][ct][j] *= sc;
    }
    asm volatile("s_wait_dscnt 0" ::: "memory");  // wave-private LDS RAW fence
    v16bf pa = load_frag_a(ps, 32);

    // row-sum of P via ones-matrix WMMA: psum[m] broadcast to all 16 lanes
    v8f psum = wmma_bf16(pa, ones, zf);
#pragma unroll
    for (int j = 0; j < 8; ++j) rs[j] += psum[j];

#pragma unroll
    for (int rw = 0; rw < 2; ++rw) {
      const bf16_t* vrow = vb + (size_t)((r0 + rw) * 64) * 2048 + kk;
#pragma unroll
      for (int ct = 0; ct < 4; ++ct)
        acc[rw][ct] =
            wmma_bf16(pa, load_frag_b(vrow + (size_t)(ct * 16) * 2048, 2048),
                      acc[rw][ct]);
    }
  }

  // -------- epilogue: normalize, comp, rule softmax, mix -----------------
  float wso[4];
#pragma unroll
  for (int ct = 0; ct < 4; ++ct) wso[ct] = Ws[QK_ + ct * 16 + nin];

  float comp[2][8];
#pragma unroll
  for (int rw = 0; rw < 2; ++rw)
#pragma unroll
    for (int j = 0; j < 8; ++j) {
      float inv = 1.f / rs[j];
      float c = 0.f;
#pragma unroll
      for (int ct = 0; ct < 4; ++ct) {
        float a = acc[rw][ct][j] * inv;
        acc[rw][ct][j] = a;
        c += a * wso[ct];
      }
      c = red16_add(c);
      comp[rw][j] = c + qvd[(size_t)(b * S_ + q0 + j + nhalf * 8) * H_ + h];
    }

  if (nin == 0) {
#pragma unroll
    for (int rw = 0; rw < 2; ++rw)
#pragma unroll
      for (int j = 0; j < 8; ++j) {
        int m = j + nhalf * 8;
        prev[((size_t)(b * S_ + q0 + m) * H_ + h) * R_ + r0 + rw] = comp[rw][j];
        sComp[qg][m][r0 + rw] = comp[rw][j];
      }
  }
  __syncthreads();

  float w0[8], w1[8];
#pragma unroll
  for (int j = 0; j < 8; ++j) {
    int m = j + nhalf * 8;
    float cv[8];
    float mx = -3e38f;
#pragma unroll
    for (int r = 0; r < 8; ++r) { cv[r] = sComp[qg][m][r]; mx = fmaxf(mx, cv[r]); }
    float sum = 0.f;
#pragma unroll
    for (int r = 0; r < 8; ++r) sum += __expf(cv[r] - mx);
    float inv = 1.f / sum;
    w0[j] = __expf(cv[r0] - mx) * inv;
    w1[j] = __expf(cv[r0 + 1] - mx) * inv;
  }
#pragma unroll
  for (int ct = 0; ct < 4; ++ct)
#pragma unroll
    for (int j = 0; j < 8; ++j) {
      int m = j + nhalf * 8;
      sO[qg][wr][m][ct * 16 + nin] =
          w0[j] * acc[0][ct][j] + w1[j] * acc[1][ct][j];
    }
  __syncthreads();

  for (int e = t; e < 2 * 16 * 64; e += 256) {
    int grp = e >> 10, m = (e >> 6) & 15, d = e & 63;
    float v = sO[grp][0][m][d] + sO[grp][1][m][d] + sO[grp][2][m][d] +
              sO[grp][3][m][d];
    int srow = blockIdx.x * 32 + grp * 16 + m;
    obuf[((size_t)(b * S_ + srow)) * DIM_ + h * 64 + d] = (bf16_t)v;
  }
}

// ---- host orchestration -------------------------------------------------
extern "C" void kernel_launch(void* const* d_in, const int* in_sizes, int n_in,
                              void* d_out, int out_size, void* d_ws,
                              size_t ws_size, hipStream_t stream) {
  (void)in_sizes; (void)n_in; (void)out_size; (void)ws_size;
  const float* x   = (const float*)d_in[0];
  const float* Wq  = (const float*)d_in[1];
  const float* bq  = (const float*)d_in[2];
  const float* Wk  = (const float*)d_in[3];
  const float* bk  = (const float*)d_in[4];
  const float* Wv  = (const float*)d_in[5];
  const float* bv  = (const float*)d_in[6];
  const float* Wqv = (const float*)d_in[7];
  const float* bqv = (const float*)d_in[8];
  const float* Ws  = (const float*)d_in[9];
  const float* bs  = (const float*)d_in[10];
  const float* Wf  = (const float*)d_in[11];
  const float* bfp = (const float*)d_in[12];
  const float* W1  = (const float*)d_in[13];
  const float* b1  = (const float*)d_in[14];
  const float* W2  = (const float*)d_in[15];
  const float* b2  = (const float*)d_in[16];
  const float* g1  = (const float*)d_in[17];
  const float* be1 = (const float*)d_in[18];
  const float* g2  = (const float*)d_in[19];
  const float* be2 = (const float*)d_in[20];

  const int M = B_ * S_;  // 8192
  char* p = (char*)d_ws;
  auto carve = [&](size_t bytes) -> void* {
    void* r = (void*)p;
    p += (bytes + 255) & ~(size_t)255;
    return r;
  };
  bf16_t* xn    = (bf16_t*)carve((size_t)M * DIM_ * 2);  // reused as "on"
  bf16_t* WcatT = (bf16_t*)carve((size_t)2432 * 768 * 2);
  float*  bcat  = (float*)carve(2432 * 4);
  bf16_t* WfT   = (bf16_t*)carve((size_t)768 * 768 * 2);
  bf16_t* W1T   = (bf16_t*)carve((size_t)1536 * 768 * 2);
  bf16_t* W2T   = (bf16_t*)carve((size_t)768 * 1536 * 2);
  bf16_t* qkv   = (bf16_t*)carve((size_t)M * 2432 * 2);  // reused as h1
  bf16_t* vTb   = (bf16_t*)carve((size_t)B_ * 512 * 2048 * 2);
  float*  qvd   = (float*)carve((size_t)B_ * S_ * H_ * 4);
  bf16_t* obuf  = (bf16_t*)carve((size_t)M * DIM_ * 2);
  float*  out1  = (float*)carve((size_t)M * DIM_ * 4);
  bf16_t* h1    = qkv;  // qkv dead after attention
  bf16_t* onb   = xn;   // xn dead after QKV gemm
  float* fout = (float*)d_out;
  float* prev = fout + (size_t)M * DIM_;

  const float sq = 0.125f;                 // 1/sqrt(HD)
  const float sqv = 0.17677669529663687f;  // 1/sqrt(QK)
  auto gKN = [](size_t kn) { return (unsigned)((kn + 255) / 256); };

  ln_kernel<<<M / 8, 256, 0, stream>>>(x, g1, be1, xn);

  packT_kernel<<<gKN((size_t)768 * 768), 256, 0, stream>>>(Wq, WcatT, 768, 768, sq);
  packT_kernel<<<gKN((size_t)768 * 768), 256, 0, stream>>>(Wk, WcatT + (size_t)768 * 768, 768, 768, 1.f);
  packT_kernel<<<gKN((size_t)768 * 512), 256, 0, stream>>>(Wv, WcatT + (size_t)1536 * 768, 768, 512, 1.f);
  packT_kernel<<<gKN((size_t)768 * 384), 256, 0, stream>>>(Wqv, WcatT + (size_t)2048 * 768, 768, 384, sqv);
  packT_kernel<<<gKN((size_t)768 * 768), 256, 0, stream>>>(Wf, WfT, 768, 768, 1.f);
  packT_kernel<<<gKN((size_t)768 * 1536), 256, 0, stream>>>(W1, W1T, 768, 1536, 1.f);
  packT_kernel<<<gKN((size_t)1536 * 768), 256, 0, stream>>>(W2, W2T, 1536, 768, 1.f);
  packb_kernel<<<3, 256, 0, stream>>>(bq, bcat, 768, sq);
  packb_kernel<<<3, 256, 0, stream>>>(bk, bcat + 768, 768, 1.f);
  packb_kernel<<<2, 256, 0, stream>>>(bv, bcat + 1536, 512, 1.f);
  packb_kernel<<<2, 256, 0, stream>>>(bqv, bcat + 2048, 384, sqv);

  // fused q|k|v|qv projection: [8192x768] @ [768x2432]
  gemm_kernel<EPI_BF16><<<dim3(2432 / BN, M / BM), 256, 0, stream>>>(
      xn, WcatT, M, 2432, 768, bcat, nullptr, qkv);

  vt_kernel<<<(B_ * 512 * 2048) / 256, 256, 0, stream>>>(qkv, vTb);
  qvdot_kernel<<<(B_ * S_ * H_ + 255) / 256, 256, 0, stream>>>(qkv, Ws, bs, qvd);

  attn_kernel<<<dim3(S_ / 32, H_, B_), 256, 0, stream>>>(qkv, vTb, qvd, Ws,
                                                         obuf, prev);

  // out1 = x + o @ Wf + bf
  gemm_kernel<EPI_F32_RES><<<dim3(768 / BN, M / BM), 256, 0, stream>>>(
      obuf, WfT, M, 768, 768, bfp, x, out1);

  ln_kernel<<<M / 8, 256, 0, stream>>>(out1, g2, be2, onb);

  gemm_kernel<EPI_BF16_RELU><<<dim3(1536 / BN, M / BM), 256, 0, stream>>>(
      onb, W1T, M, 1536, 768, b1, nullptr, h1);
  gemm_kernel<EPI_F32_RES><<<dim3(768 / BN, M / BM), 256, 0, stream>>>(
      h1, W2T, M, 768, 1536, b2, out1, fout);
}
